// EncoderBlock_71373766525163
// MI455X (gfx1250) — compile-verified, matrix operand order checked by reading, not run on hardware
//
#include <hip/hip_runtime.h>
#include <math.h>

typedef __attribute__((ext_vector_type(16))) __bf16 bf16x16;
typedef __attribute__((ext_vector_type(8)))  float  f32x8;

#define EMB   768
#define DFF   3072
#define LSEQ  1024
#define NB    16
#define NH    12
#define HD    64
#define NTOK  (NB*LSEQ)

#define EP_BF16 0
#define EP_GELU 1
#define EP_RES  2

union B128 { uint4 q; __bf16 e[8]; };
union Frag { bf16x16 v; uint4 q[2]; };

// ---------------------------------------------------------------- cast f32->bf16
__global__ void cast_f32_to_bf16(const float* __restrict__ in, __bf16* __restrict__ out, int n) {
    int i = blockIdx.x * blockDim.x + threadIdx.x;
    int stride = gridDim.x * blockDim.x;
    for (; i < n; i += stride) out[i] = (__bf16)in[i];
}

// ---------------------------------------------------------------- LayerNorm -> bf16
__global__ __launch_bounds__(256) void layernorm_bf16(const float* __restrict__ x,
                                                      const float* __restrict__ g,
                                                      const float* __restrict__ b,
                                                      __bf16* __restrict__ out) {
    __shared__ float s1[256], s2[256];
    int row = blockIdx.x, tid = threadIdx.x;
    const float* xr = x + (size_t)row * EMB;
    float a0 = xr[tid], a1 = xr[tid + 256], a2 = xr[tid + 512];
    s1[tid] = a0 + a1 + a2;
    s2[tid] = a0*a0 + a1*a1 + a2*a2;
    __syncthreads();
    for (int o = 128; o > 0; o >>= 1) {
        if (tid < o) { s1[tid] += s1[tid + o]; s2[tid] += s2[tid + o]; }
        __syncthreads();
    }
    float mu   = s1[0] * (1.0f / EMB);
    float var  = s2[0] * (1.0f / EMB) - mu * mu;
    float rstd = rsqrtf(var + 1e-5f);
    __bf16* orow = out + (size_t)row * EMB;
    orow[tid]       = (__bf16)((a0 - mu) * rstd * g[tid]       + b[tid]);
    orow[tid + 256] = (__bf16)((a1 - mu) * rstd * g[tid + 256] + b[tid + 256]);
    orow[tid + 512] = (__bf16)((a2 - mu) * rstd * g[tid + 512] + b[tid + 512]);
}

// ---------------------------------------------------------------- WMMA GEMM
// C[M,N] = A_bf16[M,K] @ B_bf16[K,N] + bias, epilogue per EPI.
// Block: 256 thr = 8 waves (2x4), block tile 64x128, wave tile 32x32, Kc=64,
// register-level prefetch of the next global tile to hide load latency.
template<int EPI>
__global__ __launch_bounds__(256) void gemm_bf16_wmma(
    const __bf16* __restrict__ A, const __bf16* __restrict__ B,
    const float* __restrict__ bias, const float* __restrict__ resid,
    void* __restrict__ outp, int M, int N, int K)
{
    __shared__ __bf16 sA[64][72];    // [m][k], row stride 144B (16B multiple)
    __shared__ __bf16 sB[128][72];   // [n][k] (transposed), fragments read 8-k runs at ko and ko+16

    int tid = threadIdx.x;
    int m0 = blockIdx.y * 64, n0 = blockIdx.x * 128;
    int wid = tid >> 5, lane = tid & 31;
    int wm = wid >> 2, wn = wid & 3;
    int lane_lo = lane & 15;
    int khalf = (lane >> 4) * 8;     // A/B fragment K sub-block per half-wave
    int rhalf = (lane >> 4) * 8;     // C-layout row offset per half-wave

    f32x8 acc[2][2] = {};

    int arow = tid >> 2, akg = (tid & 3) * 16;  // A staging: 64 rows x 4 k-groups of 16
    int brow = tid >> 2, bng = (tid & 3) * 32;  // B staging: 64 k x 4 n-groups of 32

    uint4 aReg[2];
    B128  bReg[4];
    {   // prologue: tile kb=0 into registers
        const __bf16* aP = A + (size_t)(m0 + arow) * K + akg;
        const __bf16* bP = B + (size_t)brow * N + n0 + bng;
        aReg[0] = *reinterpret_cast<const uint4*>(aP);
        aReg[1] = *reinterpret_cast<const uint4*>(aP + 8);
        #pragma unroll
        for (int u = 0; u < 4; u++) bReg[u].q = *reinterpret_cast<const uint4*>(bP + u * 8);
    }

    for (int kb = 0; kb < K; kb += 64) {
        // commit staged registers to LDS
        *reinterpret_cast<uint4*>(&sA[arow][akg])     = aReg[0];
        *reinterpret_cast<uint4*>(&sA[arow][akg + 8]) = aReg[1];
        #pragma unroll
        for (int u = 0; u < 4; u++)
            #pragma unroll
            for (int i = 0; i < 8; i++) sB[bng + u * 8 + i][brow] = bReg[u].e[i];
        __syncthreads();

        // prefetch next tile into registers (latency hidden behind WMMAs below)
        if (kb + 64 < K) {
            const __bf16* aP = A + (size_t)(m0 + arow) * K + (kb + 64) + akg;
            const __bf16* bP = B + (size_t)(kb + 64 + brow) * N + n0 + bng;
            aReg[0] = *reinterpret_cast<const uint4*>(aP);
            aReg[1] = *reinterpret_cast<const uint4*>(aP + 8);
            #pragma unroll
            for (int u = 0; u < 4; u++) bReg[u].q = *reinterpret_cast<const uint4*>(bP + u * 8);
        }

        #pragma unroll
        for (int ks = 0; ks < 2; ks++) {
            int ko = ks * 32 + khalf;
            Frag af[2], bf2[2];
            #pragma unroll
            for (int i = 0; i < 2; i++) {
                int m = wm * 32 + i * 16 + lane_lo;
                af[i].q[0] = *reinterpret_cast<const uint4*>(&sA[m][ko]);
                af[i].q[1] = *reinterpret_cast<const uint4*>(&sA[m][ko + 16]);
            }
            #pragma unroll
            for (int j = 0; j < 2; j++) {
                int n = wn * 32 + j * 16 + lane_lo;
                bf2[j].q[0] = *reinterpret_cast<const uint4*>(&sB[n][ko]);
                bf2[j].q[1] = *reinterpret_cast<const uint4*>(&sB[n][ko + 16]);
            }
            #pragma unroll
            for (int i = 0; i < 2; i++)
                #pragma unroll
                for (int j = 0; j < 2; j++)
                    acc[i][j] = __builtin_amdgcn_wmma_f32_16x16x32_bf16(
                        false, af[i].v, false, bf2[j].v, (short)0, acc[i][j], false, false);
        }
        __syncthreads();
    }

    #pragma unroll
    for (int i = 0; i < 2; i++) {
        #pragma unroll
        for (int j = 0; j < 2; j++) {
            int n_col = n0 + wn * 32 + j * 16 + lane_lo;
            float bv = bias[n_col];
            #pragma unroll
            for (int s = 0; s < 8; s++) {
                int m_row = m0 + wm * 32 + i * 16 + s + rhalf;
                size_t idx = (size_t)m_row * N + n_col;
                float val = acc[i][j][s] + bv;
                if (EPI == EP_BF16) {
                    ((__bf16*)outp)[idx] = (__bf16)val;
                } else if (EPI == EP_GELU) {
                    float gel = 0.5f * val * (1.0f + erff(val * 0.70710678118654752f));
                    ((__bf16*)outp)[idx] = (__bf16)gel;
                } else {
                    ((float*)outp)[idx] = resid[idx] + val;
                }
            }
        }
    }
}

// ---------------------------------------------------------------- Flash attention
// grid (L/64, NH, NB), 128 thr = 4 waves; wave owns 16 query rows, loop 32-key tiles
// with register-level prefetch of the next K/V tile.
__global__ __launch_bounds__(128) void attn_wmma(
    const __bf16* __restrict__ Qm, const __bf16* __restrict__ Km,
    const __bf16* __restrict__ Vm, __bf16* __restrict__ ctx)
{
    __shared__ __bf16 sK[32][72];        // [key][hd]  == B^T layout for S = Q K^T
    __shared__ __bf16 sV[64][40];        // [hd][key]  == B^T layout for O += P V
    __shared__ __bf16 sP[4][16][40];     // per-wave P scratch (C-layout -> A-frag)

    int tid = threadIdx.x;
    int wid = tid >> 5, lane = tid & 31;
    int lane_lo = lane & 15;
    int khalf = (lane >> 4) * 8;
    int rhalf = (lane >> 4) * 8;
    int b = blockIdx.z, h = blockIdx.y, rb = blockIdx.x;
    size_t tokbase = (size_t)b * LSEQ;

    // Q fragments (held in registers; reused across all key tiles)
    int qrow = rb * 64 + wid * 16 + lane_lo;
    const __bf16* qp = Qm + (tokbase + qrow) * EMB + h * HD;
    Frag qf[2];
    #pragma unroll
    for (int ks = 0; ks < 2; ks++) {
        qf[ks].q[0] = *reinterpret_cast<const uint4*>(qp + ks * 32 + khalf);
        qf[ks].q[1] = *reinterpret_cast<const uint4*>(qp + ks * 32 + khalf + 16);
    }

    f32x8 oacc[4] = {};
    float mrow[8], lrow[8];
    #pragma unroll
    for (int s = 0; s < 8; s++) { mrow[s] = -3.0e38f; lrow[s] = 0.0f; }

    int skey = tid >> 2, sg = (tid & 3) * 16;   // staging: 32 keys x 4 hd-groups of 16

    uint4 kReg[2];
    B128  vReg[2];
    {   // prologue: key tile kt=0
        const __bf16* kp = Km + (tokbase + skey) * EMB + h * HD + sg;
        const __bf16* vp = Vm + (tokbase + skey) * EMB + h * HD + sg;
        kReg[0] = *reinterpret_cast<const uint4*>(kp);
        kReg[1] = *reinterpret_cast<const uint4*>(kp + 8);
        vReg[0].q = *reinterpret_cast<const uint4*>(vp);
        vReg[1].q = *reinterpret_cast<const uint4*>(vp + 8);
    }

    for (int kt = 0; kt < LSEQ; kt += 32) {
        // commit staged registers: K tile [key][hd], V tile transposed [hd][key]
        *reinterpret_cast<uint4*>(&sK[skey][sg])     = kReg[0];
        *reinterpret_cast<uint4*>(&sK[skey][sg + 8]) = kReg[1];
        #pragma unroll
        for (int i = 0; i < 8; i++) { sV[sg + i][skey] = vReg[0].e[i]; sV[sg + 8 + i][skey] = vReg[1].e[i]; }
        __syncthreads();

        // prefetch next K/V tile
        if (kt + 32 < LSEQ) {
            const __bf16* kp = Km + (tokbase + kt + 32 + skey) * EMB + h * HD + sg;
            const __bf16* vp = Vm + (tokbase + kt + 32 + skey) * EMB + h * HD + sg;
            kReg[0] = *reinterpret_cast<const uint4*>(kp);
            kReg[1] = *reinterpret_cast<const uint4*>(kp + 8);
            vReg[0].q = *reinterpret_cast<const uint4*>(vp);
            vReg[1].q = *reinterpret_cast<const uint4*>(vp + 8);
        }

        // S = Q K^T  (16 rows x 32 keys)
        f32x8 sacc[2] = {};
        #pragma unroll
        for (int j = 0; j < 2; j++) {
            int nkey = j * 16 + lane_lo;
            #pragma unroll
            for (int ks = 0; ks < 2; ks++) {
                Frag kf;
                kf.q[0] = *reinterpret_cast<const uint4*>(&sK[nkey][ks * 32 + khalf]);
                kf.q[1] = *reinterpret_cast<const uint4*>(&sK[nkey][ks * 32 + khalf + 16]);
                sacc[j] = __builtin_amdgcn_wmma_f32_16x16x32_bf16(
                    false, qf[ks].v, false, kf.v, (short)0, sacc[j], false, false);
            }
        }

        // online softmax (row stats reduced within 16-lane halves; C-layout rows)
        float p0[8], p1[8];
        #pragma unroll
        for (int s = 0; s < 8; s++) {
            float v0s = sacc[0][s] * 0.125f;
            float v1s = sacc[1][s] * 0.125f;
            float mx = fmaxf(v0s, v1s);
            #pragma unroll
            for (int msk = 1; msk < 16; msk <<= 1) mx = fmaxf(mx, __shfl_xor(mx, msk, 32));
            float newm = fmaxf(mrow[s], mx);
            float alpha = __expf(mrow[s] - newm);
            mrow[s] = newm;
            float e0 = __expf(v0s - newm);
            float e1 = __expf(v1s - newm);
            p0[s] = e0; p1[s] = e1;
            float ls = e0 + e1;
            #pragma unroll
            for (int msk = 1; msk < 16; msk <<= 1) ls += __shfl_xor(ls, msk, 32);
            lrow[s] = lrow[s] * alpha + ls;
            #pragma unroll
            for (int t = 0; t < 4; t++) oacc[t][s] *= alpha;
        }

        // convert P: C-layout -> A-fragment via wave-private LDS scratch
        #pragma unroll
        for (int s = 0; s < 8; s++) {
            int r = s + rhalf;
            sP[wid][r][lane_lo]      = (__bf16)p0[s];
            sP[wid][r][16 + lane_lo] = (__bf16)p1[s];
        }
        asm volatile("s_wait_dscnt 0" ::: "memory");   // wave-local LDS RAW ordering
        Frag pf;
        pf.q[0] = *reinterpret_cast<const uint4*>(&sP[wid][lane_lo][khalf]);
        pf.q[1] = *reinterpret_cast<const uint4*>(&sP[wid][lane_lo][khalf + 16]);

        // O += P V  (16 rows x 64 hd)
        #pragma unroll
        for (int t = 0; t < 4; t++) {
            int nhd = t * 16 + lane_lo;
            Frag vf;
            vf.q[0] = *reinterpret_cast<const uint4*>(&sV[nhd][khalf]);
            vf.q[1] = *reinterpret_cast<const uint4*>(&sV[nhd][khalf + 16]);
            oacc[t] = __builtin_amdgcn_wmma_f32_16x16x32_bf16(
                false, pf.v, false, vf.v, (short)0, oacc[t], false, false);
        }
        __syncthreads();
    }

    // normalize and store ctx (bf16, [token][head*64+hd])
    #pragma unroll
    for (int s = 0; s < 8; s++) {
        float inv = 1.0f / lrow[s];
        int row = rb * 64 + wid * 16 + s + rhalf;
        __bf16* cp = ctx + (tokbase + row) * EMB + h * HD;
        #pragma unroll
        for (int t = 0; t < 4; t++)
            cp[t * 16 + lane_lo] = (__bf16)(oacc[t][s] * inv);
    }
}

// ---------------------------------------------------------------- host orchestration
extern "C" void kernel_launch(void* const* d_in, const int* in_sizes, int n_in,
                              void* d_out, int out_size, void* d_ws, size_t ws_size,
                              hipStream_t stream)
{
    (void)in_sizes; (void)n_in; (void)out_size; (void)ws_size;
    const float* x    = (const float*)d_in[0];
    const float* ln1g = (const float*)d_in[1];
    const float* ln1b = (const float*)d_in[2];
    const float* Wq   = (const float*)d_in[3];
    const float* bq   = (const float*)d_in[4];
    const float* Wk   = (const float*)d_in[5];
    const float* bk   = (const float*)d_in[6];
    const float* Wv   = (const float*)d_in[7];
    const float* bv   = (const float*)d_in[8];
    const float* Wo   = (const float*)d_in[9];
    const float* bo   = (const float*)d_in[10];
    const float* ln2g = (const float*)d_in[11];
    const float* ln2b = (const float*)d_in[12];
    const float* W1   = (const float*)d_in[13];
    const float* b1   = (const float*)d_in[14];
    const float* W2   = (const float*)d_in[15];
    const float* b2   = (const float*)d_in[16];

    char* ws = (char*)d_ws;
    size_t off = 0;
    auto alloc = [&](size_t bytes) -> void* {
        void* p = ws + off;
        off += (bytes + 255) & ~(size_t)255;
        return p;
    };

    const size_t actB = (size_t)NTOK * EMB * sizeof(__bf16);     // 25.17 MB
    const size_t wB   = (size_t)EMB * EMB * sizeof(__bf16);
    const size_t w1B  = (size_t)EMB * DFF * sizeof(__bf16);

    __bf16* hbf  = (__bf16*)alloc(actB);        // LN1 output, reused for LN2
    __bf16* Wqb  = (__bf16*)alloc(wB);
    __bf16* Wkb  = (__bf16*)alloc(wB);
    __bf16* Wvb  = (__bf16*)alloc(wB);
    __bf16* Wob  = (__bf16*)alloc(wB);
    __bf16* W1b  = (__bf16*)alloc(w1B);
    __bf16* W2b  = (__bf16*)alloc(w1B);
    float*  x1   = (float*)alloc((size_t)NTOK * EMB * sizeof(float));
    // Q/K/V/ctx region aliases the MLP intermediate (4*NTOK*EMB*2 == NTOK*DFF*2)
    char* region = (char*)alloc(4 * actB);
    __bf16* qb   = (__bf16*)(region);
    __bf16* kbuf = (__bf16*)(region + actB);
    __bf16* vbuf = (__bf16*)(region + 2 * actB);
    __bf16* ctx  = (__bf16*)(region + 3 * actB);
    __bf16* hmid = (__bf16*)(region);           // 16384 x 3072 bf16

    // 1) weights -> bf16
    cast_f32_to_bf16<<<dim3(1024), 256, 0, stream>>>(Wq, Wqb, EMB * EMB);
    cast_f32_to_bf16<<<dim3(1024), 256, 0, stream>>>(Wk, Wkb, EMB * EMB);
    cast_f32_to_bf16<<<dim3(1024), 256, 0, stream>>>(Wv, Wvb, EMB * EMB);
    cast_f32_to_bf16<<<dim3(1024), 256, 0, stream>>>(Wo, Wob, EMB * EMB);
    cast_f32_to_bf16<<<dim3(2048), 256, 0, stream>>>(W1, W1b, EMB * DFF);
    cast_f32_to_bf16<<<dim3(2048), 256, 0, stream>>>(W2, W2b, EMB * DFF);

    dim3 g768(EMB / 128, NTOK / 64);    // (6, 256)
    dim3 g3072(DFF / 128, NTOK / 64);   // (24, 256)

    // 2) LN1 -> h ; QKV projections
    layernorm_bf16<<<NTOK, 256, 0, stream>>>(x, ln1g, ln1b, hbf);
    gemm_bf16_wmma<EP_BF16><<<g768, 256, 0, stream>>>(hbf, Wqb, bq, nullptr, qb, NTOK, EMB, EMB);
    gemm_bf16_wmma<EP_BF16><<<g768, 256, 0, stream>>>(hbf, Wkb, bk, nullptr, kbuf, NTOK, EMB, EMB);
    gemm_bf16_wmma<EP_BF16><<<g768, 256, 0, stream>>>(hbf, Wvb, bv, nullptr, vbuf, NTOK, EMB, EMB);

    // 3) attention
    attn_wmma<<<dim3(LSEQ / 64, NH, NB), 128, 0, stream>>>(qb, kbuf, vbuf, ctx);

    // 4) output projection + residual -> x1
    gemm_bf16_wmma<EP_RES><<<g768, 256, 0, stream>>>(ctx, Wob, bo, x, x1, NTOK, EMB, EMB);

    // 5) LN2 -> h ; MLP with exact-erf GELU ; residual -> d_out
    layernorm_bf16<<<NTOK, 256, 0, stream>>>(x1, ln2g, ln2b, hbf);
    gemm_bf16_wmma<EP_GELU><<<g3072, 256, 0, stream>>>(hbf, W1b, b1, nullptr, hmid, NTOK, DFF, EMB);
    gemm_bf16_wmma<EP_RES><<<g768, 256, 0, stream>>>(hmid, W2b, b2, x1, (float*)d_out, NTOK, EMB, DFF);
}
